// TrafficModel_91225105367838
// MI455X (gfx1250) — matrix operand order repaired and checked by reading, not compiled, not run on hardware
//
#include <hip/hip_runtime.h>
#include <hip/hip_bf16.h>

typedef __attribute__((ext_vector_type(16))) __bf16 v16bf;
typedef __attribute__((ext_vector_type(8)))  __bf16 v8bf;
typedef __attribute__((ext_vector_type(8)))  float  v8f;

#define B_TOT   16384
#define T_STEPS 60
#define DIN     3
#define DENC    64
#define DH      128
#define DG      512     // 4*DH, gate order i,f,g,o
#define TILE_M  16
#define NTHREADS 512

// gfx1250 has a native v_tanh_f32 transcendental; fall back to exp+rcp
__device__ __forceinline__ float fast_tanh(float x) {
#if __has_builtin(__builtin_amdgcn_tanhf)
    return __builtin_amdgcn_tanhf(x);
#else
    // tanh(x) = 1 - 2/(e^{2x}+1), single v_rcp_f32 (no IEEE div sequence)
    return fmaf(-2.0f, __builtin_amdgcn_rcpf(__expf(2.0f * x) + 1.0f), 1.0f);
#endif
}
__device__ __forceinline__ float fast_sigm(float x) {
    // sigmoid(x) = 0.5*tanh(x/2) + 0.5  -> mul + tanh + fma
    return fmaf(0.5f, fast_tanh(0.5f * x), 0.5f);
}

extern "C" __global__ __launch_bounds__(NTHREADS, 1)
void traffic_lstm_fused(const float* __restrict__ x,
                        const float* __restrict__ W_enc,
                        const float* __restrict__ b_enc,
                        const float* __restrict__ W_ih,
                        const float* __restrict__ W_hh,
                        const float* __restrict__ b_ih,
                        const float* __restrict__ b_hh,
                        const float* __restrict__ W_dec,
                        const float* __restrict__ b_dec,
                        float* __restrict__ out)
{
    // LDS budget (~208.5 KB of the 320 KB WGP pool)
    __shared__ __bf16 Whh_s[DG * DH];        // 128 KB, row-major [512][128]
    __shared__ float  xp_s[TILE_M * DG];     //  32 KB
    __shared__ float  gates_s[TILE_M * DG];  //  32 KB
    __shared__ float  enc_s[TILE_M * DENC];  //   4 KB
    __shared__ float  c_s[TILE_M * DH];      //   8 KB
    __shared__ __bf16 h_s[TILE_M * DH];      //   4 KB
    __shared__ float  Wdec_s[DH];            // 512 B

    const int tid  = threadIdx.x;
    const int wave = tid >> 5;      // 0..15
    const int lane = tid & 31;
    const int b0   = blockIdx.x * TILE_M;

    // ---------------- setup ----------------
    // W_hh -> bf16 LDS (all blocks read the same 256 KB: L2-resident)
    for (int i = tid; i < DG * DH; i += NTHREADS)
        Whh_s[i] = (__bf16)W_hh[i];

    // encoder: enc[m][e] = b_enc[e] + sum_k x[b0+m][k]*W_enc[e][k]
    for (int i = tid; i < TILE_M * DENC; i += NTHREADS) {
        const int m = i >> 6, e = i & 63;
        float v = b_enc[e];
        #pragma unroll
        for (int k = 0; k < DIN; ++k)
            v = fmaf(x[(b0 + m) * DIN + k], W_enc[e * DIN + k], v);
        enc_s[i] = v;
    }
    if (tid < DH) Wdec_s[tid] = W_dec[tid];
    for (int i = tid; i < TILE_M * DH; i += NTHREADS) {
        c_s[i] = 0.0f;
        h_s[i] = (__bf16)0.0f;
    }
    __syncthreads();

    // xp[m][r] = b_ih[r] + b_hh[r] + enc[m] . W_ih[r]
    for (int i = tid; i < TILE_M * DG; i += NTHREADS) {
        const int m = i >> 9, r = i & (DG - 1);
        float v = b_ih[r] + b_hh[r];
        const float* wr = &W_ih[r * DENC];
        const float* er = &enc_s[m * DENC];
        #pragma unroll 8
        for (int k = 0; k < DENC; ++k)
            v = fmaf(er[k], wr[k], v);
        xp_s[i] = v;
    }
    __syncthreads();

    const float bdec = b_dec[0];
    const int n16 = lane & 15;   // B/C: N column; A: M row
    const int hi  = lane >> 4;   // lane-half selector per §7.12.2 layouts

    // ---------------- recurrence ----------------
    for (int t = 0; t < T_STEPS; ++t) {
        // phase 1: gates[16][512] = xp + h(bf16) @ Whh^T(bf16), f32 accum
        #pragma unroll
        for (int q = 0; q < 2; ++q) {
            const int nt   = wave * 2 + q;     // N-tile 0..31 (gate = nt>>3)
            const int ncol = nt * 16 + n16;    // gate column for this lane
            v8f acc;
            #pragma unroll
            for (int v = 0; v < 8; ++v)        // C layout: M = v + 8*hi, N = n16
                acc[v] = xp_s[(v + hi * 8) * DG + ncol];

            #pragma unroll
            for (int kc = 0; kc < 4; ++kc) {   // K = 128 in chunks of 32
                // A fragment (h, 16x32 bf16): lane m=n16, two 8-elem chunks
                const __bf16* hrow = &h_s[n16 * DH + kc * 32];
                v8bf alo = *(const v8bf*)(hrow + hi * 16);
                v8bf ahi = *(const v8bf*)(hrow + hi * 16 + 8);
                v16bf a = __builtin_shufflevector(alo, ahi,
                    0,1,2,3,4,5,6,7,8,9,10,11,12,13,14,15);
                // B fragment (Whh^T, 32x16 bf16): lane holds Whh[row=ncol][k..k+15]
                v16bf b = *(const v16bf*)(&Whh_s[ncol * DH + kc * 32 + hi * 16]);
                acc = __builtin_amdgcn_wmma_f32_16x16x32_bf16(
                        /*neg_a=*/false, a, /*neg_b=*/false, b,
                        /*c_mod=*/(short)0, acc,
                        /*reuse_a=*/false, /*reuse_b=*/false);
            }

            // branchless activation (wave-uniform gate type):
            // gate g (nt>>3 == 2): tanh(x)              -> si=1,  so=1,  off=0
            // gates i,f,o:         sigmoid(x)=.5*tanh(.5x)+.5 -> si=.5, so=.5, off=.5
            const bool isg = ((nt >> 3) == 2);
            const float si  = isg ? 1.0f : 0.5f;
            const float so  = isg ? 1.0f : 0.5f;
            const float off = isg ? 0.0f : 0.5f;
            #pragma unroll
            for (int v = 0; v < 8; ++v) {
                const float val = fmaf(so, fast_tanh(si * acc[v]), off);
                gates_s[(v + hi * 8) * DG + ncol] = val;
            }
        }
        __syncthreads();

        // phase 2: cell update; wave w owns batch row m=w, lane covers 4 cols
        {
            const int m  = wave;
            const int j0 = lane * 4;
            float p = 0.0f;
            #pragma unroll
            for (int u = 0; u < 4; ++u) {
                const int j = j0 + u;
                const float ig = gates_s[m * DG + j];
                const float fg = gates_s[m * DG + 128 + j];
                const float gg = gates_s[m * DG + 256 + j];
                const float og = gates_s[m * DG + 384 + j];
                const float c  = fmaf(fg, c_s[m * DH + j], ig * gg);
                c_s[m * DH + j] = c;
                const float h = og * fast_tanh(c);
                h_s[m * DH + j] = (__bf16)h;
                p = fmaf(h, Wdec_s[j], p);      // decoder partial
            }
            // wave32 reduction for the rank-1 decoder
            #pragma unroll
            for (int off = 16; off > 0; off >>= 1)
                p += __shfl_xor(p, off, 32);
            if (lane == 0)
                out[(size_t)(b0 + m) * T_STEPS + t] = fast_sigm(p + bdec);
        }
        __syncthreads();
    }
}

extern "C" void kernel_launch(void* const* d_in, const int* in_sizes, int n_in,
                              void* d_out, int out_size, void* d_ws, size_t ws_size,
                              hipStream_t stream) {
    const float* x     = (const float*)d_in[0];
    const float* W_enc = (const float*)d_in[1];
    const float* b_enc = (const float*)d_in[2];
    const float* W_ih  = (const float*)d_in[3];
    const float* W_hh  = (const float*)d_in[4];
    const float* b_ih  = (const float*)d_in[5];
    const float* b_hh  = (const float*)d_in[6];
    const float* W_dec = (const float*)d_in[7];
    const float* b_dec = (const float*)d_in[8];
    float* out = (float*)d_out;

    dim3 grid(B_TOT / TILE_M);
    dim3 block(NTHREADS);
    hipLaunchKernelGGL(traffic_lstm_fused, grid, block, 0, stream,
                       x, W_enc, b_enc, W_ih, W_hh, b_ih, b_hh, W_dec, b_dec, out);
}
